// Model_39676907883821
// MI455X (gfx1250) — matrix operand order repaired and checked by reading, not compiled, not run on hardware
//
#include <hip/hip_runtime.h>

typedef __attribute__((ext_vector_type(16))) _Float16 v16h;
typedef __attribute__((ext_vector_type(8)))  _Float16 v8h;
typedef __attribute__((ext_vector_type(2)))  _Float16 v2h;
typedef __attribute__((ext_vector_type(8)))  float    v8f;

#define S_LEN 4096
#define NHEAD 16
#define DHEAD 64
#define ROWSTRIDE (NHEAD*DHEAD)   // 1024 floats per token row
#define KBLK 64
#define NITER (S_LEN/KBLK)

// raw max ops (avoid fmaxf's canonicalize self-max expansion)
__device__ __forceinline__ float fmax_raw(float a, float b) {
  float d; asm("v_max_num_f32 %0, %1, %2" : "=v"(d) : "v"(a), "v"(b)); return d;
}
__device__ __forceinline__ float fmax3_raw(float a, float b, float c) {
  float d; asm("v_max3_num_f32 %0, %1, %2, %3" : "=v"(d) : "v"(a), "v"(b), "v"(c)); return d;
}

// DPP16 permute within 16-lane rows (our WMMA halves)
template<int CTRL>
__device__ __forceinline__ float dpp_f(float x) {
  return __builtin_bit_cast(float,
    __builtin_amdgcn_update_dpp(0, __builtin_bit_cast(int, x), CTRL, 0xf, 0xf, true));
}
// butterfly over masks {1,2,7,15}: spans all 16 lanes of a row
__device__ __forceinline__ float rowmax16(float x) {
  x = fmax_raw(x, dpp_f<0xB1>(x));   // quad_perm(1,0,3,2)  : xor 1
  x = fmax_raw(x, dpp_f<0x4E>(x));   // quad_perm(2,3,0,1)  : xor 2
  x = fmax_raw(x, dpp_f<0x141>(x));  // row_half_mirror     : xor 7
  x = fmax_raw(x, dpp_f<0x140>(x));  // row_mirror          : xor 15
  return x;
}
__device__ __forceinline__ float rowsum16(float x) {
  x += dpp_f<0xB1>(x);
  x += dpp_f<0x4E>(x);
  x += dpp_f<0x141>(x);
  x += dpp_f<0x140>(x);
  return x;
}

// K-axis permutation shared by P (A side, natural packed order) and V (B side):
// position kd of a 32-key chunk holds key g(kd)
__device__ __forceinline__ int gperm(int kd) {
  return ((kd >> 1) & 3) | (((kd >> 3) & 1) << 2) | (((kd >> 4) & 1) << 3) | ((kd & 1) << 4);
}

__global__ __launch_bounds__(256) void
fa_fwd(const float* __restrict__ q, const float* __restrict__ k,
       const float* __restrict__ v, float* __restrict__ out)
{
  __shared__ _Float16 kbuf[2][64][64];    // [buf][key row][k-dim]                16 KB
  __shared__ _Float16 vtbuf[2][64][64];   // [buf][v col][key pos] (V^T, g-perm)  16 KB
  __shared__ _Float16 pbuf[8][2][8][64];  // [wave][chunk][r][2*lane] packed P    16 KB

  const int tid  = threadIdx.x;
  const int wv   = tid >> 5;
  const int lane = tid & 31;
  const int hf   = lane >> 4;
  const int l16  = lane & 15;

  const int h    = blockIdx.x & (NHEAD-1);
  const int qblk = blockIdx.x >> 4;
  const int n0   = qblk*128 + wv*16;

  // ---- Q A-fragments, prescaled by (1/sqrt(64)) * log2(e) so exp == v_exp_f32 ----
  const float QSCALE = 0.125f * 1.4426950408889634f;
  const float* qp = q + (size_t)(n0 + l16)*ROWSTRIDE + h*DHEAD;
  const int b0 = hf*16, b1 = hf*16 + 8;
  v16h a0, a1;
#pragma unroll
  for (int e = 0; e < 8; ++e) {
    a0[e]   = (_Float16)(qp[b0+e]    * QSCALE);
    a0[e+8] = (_Float16)(qp[b1+e]    * QSCALE);
    a1[e]   = (_Float16)(qp[32+b0+e] * QSCALE);
    a1[e+8] = (_Float16)(qp[32+b1+e] * QSCALE);
  }

  v8f o0 = {}, o1 = {}, o2 = {}, o3 = {};
  float mrow[8], lsum[8];          // lsum = per-lane PARTIAL sums (reduced in epilogue)
  unsigned rowkey[8];
#pragma unroll
  for (int r = 0; r < 8; ++r) {
    mrow[r] = -3.0e38f; lsum[r] = 0.f;
    rowkey[r] = ((unsigned)h << 24) | ((unsigned)(n0 + r + 8*hf) << 12);
  }

  // staging thread mapping (64x64 tiles, 256 threads -> 16 elems each)
  const int kj   = tid >> 2;           // K: key row 0..63
  const int kc   = (tid & 3) * 16;     // K: k-dim chunk base
  const int vcol = tid & 63;           // V: column
  const int vj   = (tid >> 6) * 16;    // V: key position base (0,16,32,48)
  int vrow[16];                        // g-permuted key rows this thread stages
#pragma unroll
  for (int e = 0; e < 16; ++e) {
    const int p = vj + e;
    vrow[e] = (p & 32) | gperm(p & 31);
  }
  float kreg[16], vreg[16];

#pragma unroll
  for (int e = 0; e < 16; ++e) {
    kreg[e] = k[(size_t)kj*ROWSTRIDE + h*DHEAD + kc + e];
    vreg[e] = v[(size_t)vrow[e]*ROWSTRIDE + h*DHEAD + vcol];
  }

  for (int it = 0; it < NITER; ++it) {
    const int buf = it & 1;
    const int j0  = it * KBLK;

    // ---- stage current 64-key block into LDS (f32 -> f16) ----
    {
      v8h kd0, kd1, vd0, vd1;
#pragma unroll
      for (int e = 0; e < 8; ++e) {
        kd0[e] = (_Float16)kreg[e];   kd1[e] = (_Float16)kreg[8+e];
        vd0[e] = (_Float16)vreg[e];   vd1[e] = (_Float16)vreg[8+e];
      }
      *(v8h*)&kbuf[buf][kj][kc]       = kd0;
      *(v8h*)&kbuf[buf][kj][kc+8]     = kd1;
      *(v8h*)&vtbuf[buf][vcol][vj]    = vd0;
      *(v8h*)&vtbuf[buf][vcol][vj+8]  = vd1;
    }
    __syncthreads();

    // ---- prefetch next block (overlaps all compute below) ----
    if (it + 1 < NITER) {
      const int jn = j0 + KBLK;
#pragma unroll
      for (int e = 0; e < 16; ++e) {
        kreg[e] = k[(size_t)(jn + kj)*ROWSTRIDE + h*DHEAD + kc + e];
        vreg[e] = v[(size_t)(jn + vrow[e])*ROWSTRIDE + h*DHEAD + vcol];
      }
    }

    // ---- S = (Q * qscale) K^T : four 16-key tiles, 8 WMMAs ----
    v8f s0 = {}, s1 = {}, s2 = {}, s3 = {};
    {
      v16h bka, bkb;
#define QK_TILE(SREG, T) \
      bka = *(const v16h*)&kbuf[buf][16*(T) + l16][     hf*16]; \
      bkb = *(const v16h*)&kbuf[buf][16*(T) + l16][32 + hf*16]; \
      SREG = __builtin_amdgcn_wmma_f32_16x16x32_f16(false, a0, false, bka, (short)0, SREG, false, false); \
      SREG = __builtin_amdgcn_wmma_f32_16x16x32_f16(false, a1, false, bkb, (short)0, SREG, false, false);
      QK_TILE(s0, 0) QK_TILE(s1, 1) QK_TILE(s2, 2) QK_TILE(s3, 3)
#undef QK_TILE
    }

    // ---- online softmax in exp2 domain (lane-partial row sums) ----
    float alpha[8];
    v8f p0, p1, p2, p3;
#pragma unroll
    for (int r = 0; r < 8; ++r) {
      float rm = fmax3_raw(s0[r], s1[r], s2[r]);
      rm = rowmax16(fmax3_raw(rm, s3[r], mrow[r]));
      // rm >= mrow[r] in every lane (mrow was row-uniform), and rm is row-uniform
      alpha[r] = __builtin_amdgcn_exp2f(mrow[r] - rm);
      mrow[r]  = rm;
      p0[r] = __builtin_amdgcn_exp2f(s0[r] - rm);
      p1[r] = __builtin_amdgcn_exp2f(s1[r] - rm);
      p2[r] = __builtin_amdgcn_exp2f(s2[r] - rm);
      p3[r] = __builtin_amdgcn_exp2f(s3[r] - rm);
      lsum[r] = lsum[r]*alpha[r] + ((p0[r] + p1[r]) + (p2[r] + p3[r]));
      o0[r] *= alpha[r]; o1[r] *= alpha[r]; o2[r] *= alpha[r]; o3[r] *= alpha[r];
    }

    // ---- dropout (1 hash -> 4 key-columns) + packed P store (pair = tiles t,t+1) ----
#pragma unroll
    for (int r = 0; r < 8; ++r) {
      unsigned x = rowkey[r] | (unsigned)(j0 + l16);
      x *= 2654435761u; x ^= x >> 13; x *= 0x5bd1e995u;
      const float pe0 = ((( x        & 255u) >= 26u) ? p0[r] : 0.f);
      const float pe1 = ((((x >>  8) & 255u) >= 26u) ? p1[r] : 0.f);
      const float pe2 = ((((x >> 16) & 255u) >= 26u) ? p2[r] : 0.f);
      const float pe3 = ((( x >> 24        ) >= 26u) ? p3[r] : 0.f);
      v2h pk01, pk23;
      pk01[0] = (_Float16)pe0; pk01[1] = (_Float16)pe1;   // keys (j0+c, j0+16+c)
      pk23[0] = (_Float16)pe2; pk23[1] = (_Float16)pe3;   // keys (j0+32+c, j0+48+c)
      *(v2h*)&pbuf[wv][0][r][2*lane] = pk01;
      *(v2h*)&pbuf[wv][1][r][2*lane] = pk23;
    }
    asm volatile("s_wait_dscnt 0" ::: "memory");  // same-wave DS in-order transpose fence

    // ---- P A-fragments: contiguous pair-interleaved reads; K-order = gperm,
    //      which matches the g-permuted vtbuf ----
    const int poff = 2*(hf*8 + ((l16 >> 3) & 1)*16);
    v16h pa0 = *(const v16h*)(&pbuf[wv][0][l16 & 7][0] + poff);
    v16h pa1 = *(const v16h*)(&pbuf[wv][1][l16 & 7][0] + poff);

    // ---- O += P V : 8 WMMAs (two 32-key chunks x four 16-col chunks) ----
    {
      v16h vba, vbb;
#define PV_TILE(OREG, CC) \
      vba = *(const v16h*)&vtbuf[buf][16*(CC) + l16][     hf*16]; \
      vbb = *(const v16h*)&vtbuf[buf][16*(CC) + l16][32 + hf*16]; \
      OREG = __builtin_amdgcn_wmma_f32_16x16x32_f16(false, pa0, false, vba, (short)0, OREG, false, false); \
      OREG = __builtin_amdgcn_wmma_f32_16x16x32_f16(false, pa1, false, vbb, (short)0, OREG, false, false);
      PV_TILE(o0, 0) PV_TILE(o1, 1) PV_TILE(o2, 2) PV_TILE(o3, 3)
#undef PV_TILE
    }
  }

  // ---- epilogue: reduce lane-partial sums once, normalize, permuted store ----
  float inv[8];
#pragma unroll
  for (int r = 0; r < 8; ++r) inv[r] = 1.0f / (0.9f * rowsum16(lsum[r]));
#pragma unroll
  for (int r = 0; r < 8; ++r) {
    const int qn = n0 + r + 8*hf;
    const unsigned t = (unsigned)qn*16u + (unsigned)h;   // reference reshape permutation
    const size_t base = (size_t)(t & 4095u)*ROWSTRIDE + (size_t)(t >> 12)*DHEAD;
    out[base +  0 + l16] = o0[r]*inv[r];
    out[base + 16 + l16] = o1[r]*inv[r];
    out[base + 32 + l16] = o2[r]*inv[r];
    out[base + 48 + l16] = o3[r]*inv[r];
  }
}

extern "C" void kernel_launch(void* const* d_in, const int* in_sizes, int n_in,
                              void* d_out, int out_size, void* d_ws, size_t ws_size,
                              hipStream_t stream) {
  (void)in_sizes; (void)n_in; (void)d_ws; (void)ws_size; (void)out_size;
  const float* q = (const float*)d_in[0];
  const float* k = (const float*)d_in[1];
  const float* v = (const float*)d_in[2];
  float* out = (float*)d_out;
  dim3 grid(NHEAD * (S_LEN/128));   // 512 blocks: 16 heads x 32 query blocks
  dim3 block(256);                  // 8 wave32 per block
  hipLaunchKernelGGL(fa_fwd, grid, block, 0, stream, q, k, v, out);
}
